// ProbSparseMultiheadAttention_53867479827112
// MI455X (gfx1250) — matrix operand order repaired and checked by reading, not compiled, not run on hardware
//
#include <hip/hip_runtime.h>
#include <hip/hip_bf16.h>
#include <stdint.h>

// ---------------- problem constants ----------------
#define BATCH 4
#define SEQ   2048
#define DM    512
#define NH    8
#define DK    64
#define BH    (BATCH * NH)      // 32
#define ROWS  (BATCH * SEQ)     // 8192
#define NTOP  40
#define NSAMP 40
#define RPAD  48                // 40 padded to 3 WMMA tiles

typedef __bf16 bf16;
typedef __attribute__((ext_vector_type(16))) __bf16 v16bf;
typedef __attribute__((ext_vector_type(8)))  __bf16 v8bf;
typedef __attribute__((ext_vector_type(4)))  __bf16 v4bf;
typedef __attribute__((ext_vector_type(8)))  float  v8f;
typedef int v4i __attribute__((__vector_size__(4 * sizeof(int))));

// ---------------- scalar helpers ----------------
__device__ __forceinline__ bf16 f2bf(float f) {
    union { float f; uint32_t u; } x; x.f = f;
    uint32_t r = (x.u + 0x7FFFu + ((x.u >> 16) & 1u)) >> 16;   // RNE
    uint16_t h = (uint16_t)r;
    return __builtin_bit_cast(bf16, h);
}
__device__ __forceinline__ float bf2f(bf16 b) {
    uint16_t h = __builtin_bit_cast(uint16_t, b);
    union { uint32_t u; float f; } x; x.u = ((uint32_t)h) << 16;
    return x.f;
}
// two packed bf16 in a uint32 -> floats
__device__ __forceinline__ float bflo(uint32_t u) {
    union { uint32_t u; float f; } x; x.u = u << 16; return x.f;
}
__device__ __forceinline__ float bfhi(uint32_t u) {
    union { uint32_t u; float f; } x; x.u = u & 0xFFFF0000u; return x.f;
}
__device__ __forceinline__ unsigned ihash(unsigned a) {
    a ^= a >> 16; a *= 0x7feb352du; a ^= a >> 15; a *= 0x846ca68bu; a ^= a >> 16;
    return a;
}

// ---------------- fragment loaders (pure b128 traffic) ----------------
// A-fragment (this file's k order, not the ISA table order): lane's 16 elements are K = kbase+0..15
// (contiguous) -> two contiguous 16-byte chunks.
__device__ __forceinline__ v16bf load_afrag(const bf16* __restrict__ ap, int kbase) {
    v8bf lo = *(const v8bf*)(ap + kbase);
    v8bf hi = *(const v8bf*)(ap + kbase + 8);
    return __builtin_shufflevector(lo, hi, 0, 1, 2, 3, 4, 5, 6, 7,
                                           8, 9, 10, 11, 12, 13, 14, 15);
}
// B-fragment: lane's 16 elements are 16 contiguous K values.
__device__ __forceinline__ v16bf load_bfrag(const bf16* __restrict__ bp) {
    v8bf lo = *(const v8bf*)(bp);
    v8bf hi = *(const v8bf*)(bp + 8);
    return __builtin_shufflevector(lo, hi, 0, 1, 2, 3, 4, 5, 6, 7,
                                           8, 9, 10, 11, 12, 13, 14, 15);
}
// A-fragment from LDS (emits ds_load_b128 x2)
__device__ __forceinline__ v16bf load_afrag_lds(const bf16* ap, int kbase) {
    v8bf lo = *(const v8bf*)(ap + kbase);
    v8bf hi = *(const v8bf*)(ap + kbase + 8);
    return __builtin_shufflevector(lo, hi, 0, 1, 2, 3, 4, 5, 6, 7,
                                           8, 9, 10, 11, 12, 13, 14, 15);
}
// gfx1250 async global->LDS copy of 16 bytes (GLOBAL_LOAD_ASYNC_TO_LDS_B128)
__device__ __forceinline__ void async_copy_b128(const bf16* g, bf16* l) {
    __builtin_amdgcn_global_load_async_to_lds_b128(
        (__attribute__((address_space(1))) v4i*)(v4i*)(void*)const_cast<bf16*>(g),
        (__attribute__((address_space(3))) v4i*)(v4i*)(void*)l,
        0, 0);
}

// ================================================================
// Weight transpose+convert: W[512K x 512N] f32 -> Wt[512N x 512K] bf16
// ================================================================
__global__ void transpose_w(const float* __restrict__ W, bf16* __restrict__ Wt) {
    __shared__ float t[32][33];
    const int k0 = blockIdx.x * 32, n0 = blockIdx.y * 32;
    const int tx = threadIdx.x, ty = threadIdx.y;        // (32, 8)
    #pragma unroll
    for (int i = 0; i < 32; i += 8)
        t[ty + i][tx] = W[(size_t)(k0 + ty + i) * 512 + n0 + tx];
    __syncthreads();
    #pragma unroll
    for (int i = 0; i < 32; i += 8)
        Wt[(size_t)(n0 + ty + i) * 512 + k0 + tx] = f2bf(t[tx][ty + i]);
}

// ================================================================
// Activation convert: f32 -> bf16, 4 elements / thread (b128 in, b64 out)
// ================================================================
__global__ void f32_to_bf16(const float* __restrict__ src, bf16* __restrict__ dst, int n) {
    const int i = (blockIdx.x * blockDim.x + threadIdx.x) * 4;
    if (i >= n) return;
    const float4 v = *(const float4*)(src + i);
    v4bf o = { f2bf(v.x), f2bf(v.y), f2bf(v.z), f2bf(v.w) };
    *(v4bf*)(dst + i) = o;
}

// ================================================================
// Generic 512-wide GEMM (all-bf16 operands):
//   out[r,c] = A[r,:512] @ Wt[c,:512] + bias[c]
// The block's 16x512 A tile is staged in LDS once via async-to-LDS
// (all 8 waves share it); B fragments stream from global (Wt is 0.5 MB,
// L2/WGP$ resident). One wave = 16x64 strip, K-loop step 32.
// ================================================================
__global__ void gemm512_wmma(const bf16* __restrict__ A,
                             const bf16* __restrict__ Wt,
                             const float* __restrict__ bias,
                             float* __restrict__ out_f32,
                             bf16*  __restrict__ out_qkv,
                             int vtrans, int R) {
    __shared__ alignas(16) bf16 atile[16 * 512];         // 16 KB
    const int lane = threadIdx.x & 31;
    const int wave = threadIdx.x >> 5;
    const int m0 = blockIdx.x * 16;          // gridDim.x == R/16
    const int n0 = wave * 64;                // 8 waves cover 512 cols

    // ---- async stage A tile (16 rows x 512 cols, contiguous 16 KB) ----
    {
        const bf16* gsrc = A + (size_t)m0 * 512;
        #pragma unroll
        for (int i = 0; i < 4; ++i) {
            const int off = (threadIdx.x + i * 256) * 8; // 8 bf16 = 16 B
            async_copy_b128(gsrc + off, atile + off);
        }
        asm volatile("s_wait_asynccnt 0x0" ::: "memory");
        __syncthreads();
    }

    const int half = lane >> 4;              // 0 or 1
    const int l16  = lane & 15;
    const bf16* ap  = atile + (size_t)l16 * 512;
    const bf16* wp0 = Wt + (size_t)(n0 + l16) * 512;
    const bf16* wp1 = wp0 + (size_t)16 * 512;
    const bf16* wp2 = wp0 + (size_t)32 * 512;
    const bf16* wp3 = wp0 + (size_t)48 * 512;

    v8f acc0 = {}, acc1 = {}, acc2 = {}, acc3 = {};

    for (int k0 = 0; k0 < 512; k0 += 32) {
        const int kbase = k0 + (half << 4);
        const int kb    = k0 + (half << 4);
        v16bf af = load_afrag_lds(ap, kbase);
        acc0 = __builtin_amdgcn_wmma_f32_16x16x32_bf16(
            false, af, false, load_bfrag(wp0 + kb), (short)0, acc0, false, false);
        acc1 = __builtin_amdgcn_wmma_f32_16x16x32_bf16(
            false, af, false, load_bfrag(wp1 + kb), (short)0, acc1, false, false);
        acc2 = __builtin_amdgcn_wmma_f32_16x16x32_bf16(
            false, af, false, load_bfrag(wp2 + kb), (short)0, acc2, false, false);
        acc3 = __builtin_amdgcn_wmma_f32_16x16x32_bf16(
            false, af, false, load_bfrag(wp3 + kb), (short)0, acc3, false, false);
    }

    const float b0 = bias[n0 + l16];
    const float b1 = bias[n0 + 16 + l16];
    const float b2 = bias[n0 + 32 + l16];
    const float b3 = bias[n0 + 48 + l16];

    #pragma unroll
    for (int i = 0; i < 8; ++i) {
        const int row = m0 + i + (half << 3);
        const int b = row >> 11;             // row / SEQ
        const int l = row & (SEQ - 1);
        #pragma unroll
        for (int j = 0; j < 4; ++j) {
            const int col = n0 + j * 16 + l16;
            float v = ((j == 0) ? acc0[i] : (j == 1) ? acc1[i] : (j == 2) ? acc2[i] : acc3[i])
                      + ((j == 0) ? b0 : (j == 1) ? b1 : (j == 2) ? b2 : b3);
            if (out_f32) out_f32[(size_t)row * 512 + col] = v;
            if (out_qkv) {
                const int h = col >> 6, d = col & 63;
                const int bh = b * NH + h;
                const size_t idx = vtrans
                    ? ((size_t)bh * DK + d) * SEQ + l      // Vt[bh][d][l]
                    : ((size_t)bh * SEQ + l) * DK + d;     // Q/K[bh][l][d]
                out_qkv[idx] = f2bf(v);
            }
        }
    }
}

// ================================================================
// ProbSparse sampling score: M[bh,l] = max_s(q.k_s) - mean_s(q.k_s)
// Vectorized uint4 (8 x bf16) loads.
// ================================================================
__global__ void sample_scores(const bf16* __restrict__ Q,
                              const bf16* __restrict__ K,
                              float* __restrict__ M) {
    const int t = blockIdx.x * blockDim.x + threadIdx.x;
    if (t >= BH * SEQ) return;
    const int bh = t / SEQ, l = t % SEQ;

    float qv[DK];
    const uint4* qp = (const uint4*)(Q + ((size_t)bh * SEQ + l) * DK);
    #pragma unroll
    for (int w = 0; w < 8; ++w) {
        const uint4 u = qp[w];
        qv[w * 8 + 0] = bflo(u.x); qv[w * 8 + 1] = bfhi(u.x);
        qv[w * 8 + 2] = bflo(u.y); qv[w * 8 + 3] = bfhi(u.y);
        qv[w * 8 + 4] = bflo(u.z); qv[w * 8 + 5] = bfhi(u.z);
        qv[w * 8 + 6] = bflo(u.w); qv[w * 8 + 7] = bfhi(u.w);
    }

    float mx = -3.4e38f, sum = 0.f;
    for (int s = 0; s < NSAMP; ++s) {
        const int idx = ihash((unsigned)(l * NSAMP + s) * 2654435769u + 12345u) & (SEQ - 1);
        const uint4* kp = (const uint4*)(K + ((size_t)bh * SEQ + idx) * DK);
        float acc = 0.f;
        #pragma unroll
        for (int w = 0; w < 8; ++w) {
            const uint4 u = kp[w];
            acc += qv[w * 8 + 0] * bflo(u.x) + qv[w * 8 + 1] * bfhi(u.x);
            acc += qv[w * 8 + 2] * bflo(u.y) + qv[w * 8 + 3] * bfhi(u.y);
            acc += qv[w * 8 + 4] * bflo(u.z) + qv[w * 8 + 5] * bfhi(u.z);
            acc += qv[w * 8 + 6] * bflo(u.w) + qv[w * 8 + 7] * bfhi(u.w);
        }
        mx = fmaxf(mx, acc);
        sum += acc;
    }
    M[t] = mx - sum * (1.0f / NSAMP);
}

// ================================================================
// Top-40 argmax per (b,h): one 256-thread block per bh.
// ================================================================
__global__ void topk40(const float* __restrict__ M, int* __restrict__ top) {
    __shared__ float sm[SEQ];
    __shared__ float sval[256];
    __shared__ int   sidx[256];
    const int tid = threadIdx.x, bh = blockIdx.x;
    for (int i = tid; i < SEQ; i += 256) sm[i] = M[(size_t)bh * SEQ + i];
    __syncthreads();
    for (int t = 0; t < NTOP; ++t) {
        float best = -3.4e38f; int bi = 0;
        for (int i = tid; i < SEQ; i += 256) {
            float v = sm[i];
            if (v > best) { best = v; bi = i; }
        }
        sval[tid] = best; sidx[tid] = bi;
        __syncthreads();
        for (int s = 128; s > 0; s >>= 1) {
            if (tid < s && sval[tid + s] > sval[tid]) {
                sval[tid] = sval[tid + s]; sidx[tid] = sidx[tid + s];
            }
            __syncthreads();
        }
        if (tid == 0) { top[bh * NTOP + t] = sidx[0]; sm[sidx[0]] = -3.4e38f; }
        __syncthreads();
    }
}

// ================================================================
// scores[bh, 48, 2048] = (q_red @ K^T) / sqrt(Dk)
// One wave per (bh, key-tile of 16): 3 M-tiles x 2 wmma steps.
// ================================================================
__global__ void qk_scores_wmma(const bf16* __restrict__ Q,
                               const bf16* __restrict__ K,
                               const int*  __restrict__ top,
                               float* __restrict__ scores) {
    const int gw = (blockIdx.x * blockDim.x + threadIdx.x) >> 5;
    if (gw >= BH * (SEQ / 16)) return;
    const int lane = threadIdx.x & 31;
    const int bh = gw / (SEQ / 16);
    const int kt = gw % (SEQ / 16);
    const int half = lane >> 4, l16 = lane & 15;
    const bf16* kp = K + ((size_t)bh * SEQ + kt * 16 + l16) * DK;

    #pragma unroll
    for (int mt = 0; mt < 3; ++mt) {
        const int u = mt * 16 + l16;                      // reduced-query row
        const int srcl = (u < NTOP) ? top[bh * NTOP + u] : 0;
        const bf16* qp = Q + ((size_t)bh * SEQ + srcl) * DK;
        v8f acc = {};
        #pragma unroll
        for (int kk = 0; kk < DK; kk += 32) {
            v16bf af = load_afrag(qp, kk + (half << 4));
            if (u >= NTOP) af = (v16bf)(__bf16)0.0f;      // zero padding rows
            v16bf bfr = load_bfrag(kp + kk + (half << 4));
            acc = __builtin_amdgcn_wmma_f32_16x16x32_bf16(
                false, af, false, bfr, (short)0, acc, false, false);
        }
        const float scale = 0.125f;                       // 1/sqrt(64)
        #pragma unroll
        for (int i = 0; i < 8; ++i) {
            const int row = mt * 16 + i + (half << 3);
            scores[((size_t)bh * RPAD + row) * SEQ + kt * 16 + l16] = acc[i] * scale;
        }
    }
}

// ================================================================
// Row softmax over 2048, write attn as bf16 (padding rows -> 0).
// One wave per row.
// ================================================================
__global__ void softmax_rows(const float* __restrict__ scores,
                             bf16* __restrict__ attn) {
    const int gw = (blockIdx.x * blockDim.x + threadIdx.x) >> 5;
    if (gw >= BH * RPAD) return;
    const int lane = threadIdx.x & 31;
    const int bh = gw / RPAD, u = gw % RPAD;
    const float* srow = scores + ((size_t)bh * RPAD + u) * SEQ;
    bf16* arow = attn + ((size_t)bh * RPAD + u) * SEQ;
    if (u >= NTOP) {
        const v4bf z = {};
        for (int i = lane * 4; i < SEQ; i += 128) *(v4bf*)(arow + i) = z;
        return;
    }
    float m = -3.4e38f;
    for (int i = lane; i < SEQ; i += 32) m = fmaxf(m, srow[i]);
    #pragma unroll
    for (int o = 16; o > 0; o >>= 1) m = fmaxf(m, __shfl_xor(m, o, 32));
    float s = 0.f;
    for (int i = lane; i < SEQ; i += 32) s += __expf(srow[i] - m);
    #pragma unroll
    for (int o = 16; o > 0; o >>= 1) s += __shfl_xor(s, o, 32);
    const float inv = 1.0f / s;
    for (int i = lane * 2; i < SEQ; i += 64) {
        arow[i]     = f2bf(__expf(srow[i] - m) * inv);
        arow[i + 1] = f2bf(__expf(srow[i + 1] - m) * inv);
    }
}

// ================================================================
// out_top[bh, 48, 64] = attn[48x2048] @ V[2048x64]  (V stored transposed)
// One wave per (bh, mtile, ntile); K-loop 2048 step 32.
// ================================================================
__global__ void av_wmma(const bf16* __restrict__ attn,
                        const bf16* __restrict__ Vt,   // [bh][dk][SEQ]
                        float* __restrict__ outtop) {
    const int gw = (blockIdx.x * blockDim.x + threadIdx.x) >> 5;
    if (gw >= BH * 3 * 4) return;
    const int lane = threadIdx.x & 31;
    const int bh = gw / 12;
    const int mt = (gw % 12) / 4;
    const int nt = gw % 4;
    const int half = lane >> 4, l16 = lane & 15;

    const bf16* arow = attn + ((size_t)bh * RPAD + mt * 16 + l16) * SEQ;
    const bf16* vrow = Vt + ((size_t)bh * DK + nt * 16 + l16) * SEQ;
    v8f acc = {};
    for (int k0 = 0; k0 < SEQ; k0 += 32) {
        v16bf af  = load_afrag(arow, k0 + (half << 4));
        v16bf bfr = load_bfrag(vrow + k0 + (half << 4));
        acc = __builtin_amdgcn_wmma_f32_16x16x32_bf16(
            false, af, false, bfr, (short)0, acc, false, false);
    }
    #pragma unroll
    for (int i = 0; i < 8; ++i) {
        const int row = mt * 16 + i + (half << 3);
        outtop[((size_t)bh * RPAD + row) * DK + nt * 16 + l16] = acc[i];
    }
}

// ================================================================
// vmean[bh, d] = mean_l Vt[bh, d, l]  — one wave per (bh,d), contiguous row.
// ================================================================
__global__ void vmean_kernel(const bf16* __restrict__ Vt, float* __restrict__ vmean) {
    const int gw = (blockIdx.x * blockDim.x + threadIdx.x) >> 5;
    if (gw >= BH * DK) return;
    const int lane = threadIdx.x & 31;
    const uint32_t* p = (const uint32_t*)(Vt + (size_t)gw * SEQ);
    float s = 0.f;
    for (int i = lane; i < SEQ / 2; i += 32) {
        const uint32_t u = p[i];
        s += bflo(u) + bfhi(u);
    }
    #pragma unroll
    for (int o = 16; o > 0; o >>= 1) s += __shfl_xor(s, o, 32);
    if (lane == 0) vmean[gw] = s * (1.0f / SEQ);
}

// ctx (bf16) [b,l,c] = vmean broadcast; 4 cols / thread
__global__ void ctx_fill(const float* __restrict__ vmean, bf16* __restrict__ ctx) {
    const size_t i = ((size_t)blockIdx.x * blockDim.x + threadIdx.x) * 4;
    if (i >= (size_t)ROWS * DM) return;
    const int c = (int)(i % DM);
    const int b = (int)(i / ((size_t)SEQ * DM));
    const float* vm = vmean + (b * NH + (c >> 6)) * DK + (c & 63);
    v4bf o = { f2bf(vm[0]), f2bf(vm[1]), f2bf(vm[2]), f2bf(vm[3]) };
    *(v4bf*)(ctx + i) = o;
}

// scatter the 40 attended rows into ctx (bf16)
__global__ void ctx_scatter(const int* __restrict__ top,
                            const float* __restrict__ outtop,
                            bf16* __restrict__ ctx) {
    const int t = blockIdx.x * blockDim.x + threadIdx.x;
    if (t >= BH * NTOP * DK) return;
    const int d = t & 63;
    const int u = (t >> 6) % NTOP;
    const int bh = t / (NTOP * DK);
    const int b = bh / NH, h = bh % NH;
    const int l = top[bh * NTOP + u];
    ctx[((size_t)(b * SEQ + l)) * DM + h * DK + d] =
        f2bf(outtop[((size_t)bh * RPAD + u) * DK + d]);
}

// ================================================================
extern "C" void kernel_launch(void* const* d_in, const int* in_sizes, int n_in,
                              void* d_out, int out_size, void* d_ws, size_t ws_size,
                              hipStream_t stream) {
    const float* query = (const float*)d_in[0];
    const float* key   = (const float*)d_in[1];
    const float* value = (const float*)d_in[2];
    const float* Wq = (const float*)d_in[3];  const float* bq = (const float*)d_in[4];
    const float* Wk = (const float*)d_in[5];  const float* bk = (const float*)d_in[6];
    const float* Wv = (const float*)d_in[7];  const float* bv = (const float*)d_in[8];
    const float* Wo = (const float*)d_in[9];  const float* bo = (const float*)d_in[10];
    float* out = (float*)d_out;

    // ---- workspace carve-up (256B aligned) ----
    uint8_t* ws = (uint8_t*)d_ws;
    size_t off = 0;
    auto alloc = [&](size_t bytes) -> void* {
        void* p = ws + off;
        off += (bytes + 255) & ~(size_t)255;
        return p;
    };
    bf16*  Wtq = (bf16*)alloc((size_t)DM * DM * sizeof(bf16));   // 512 KB each
    bf16*  Wtk = (bf16*)alloc((size_t)DM * DM * sizeof(bf16));
    bf16*  Wtv = (bf16*)alloc((size_t)DM * DM * sizeof(bf16));
    bf16*  Wto = (bf16*)alloc((size_t)DM * DM * sizeof(bf16));
    bf16*  Qin = (bf16*)alloc((size_t)ROWS * DM * sizeof(bf16)); // 8 MB
    bf16*  Kin = (bf16*)alloc((size_t)ROWS * DM * sizeof(bf16)); // 8 MB
    bf16*  Vin = (bf16*)alloc((size_t)ROWS * DM * sizeof(bf16)); // 8 MB
    bf16*  Qb  = (bf16*)alloc((size_t)BH * SEQ * DK * sizeof(bf16)); // 8 MB
    bf16*  Kb  = (bf16*)alloc((size_t)BH * SEQ * DK * sizeof(bf16)); // 8 MB
    bf16*  Vt  = (bf16*)alloc((size_t)BH * DK * SEQ * sizeof(bf16)); // 8 MB
    float* Msc = (float*)alloc((size_t)BH * SEQ * sizeof(float));
    int*   top = (int*)  alloc((size_t)BH * NTOP * sizeof(int));
    float* scores = (float*)alloc((size_t)BH * RPAD * SEQ * sizeof(float)); // 12 MB
    bf16*  attn   = (bf16*) alloc((size_t)BH * RPAD * SEQ * sizeof(bf16));  // 6 MB
    float* outtop = (float*)alloc((size_t)BH * RPAD * DK * sizeof(float));
    float* vmean  = (float*)alloc((size_t)BH * DK * sizeof(float));
    bf16*  ctxb   = Qin;   // alias: Qin fully consumed before ctx is built
    (void)ws_size; (void)in_sizes; (void)n_in; (void)out_size;

    const dim3 blk256(256);
    const int nAct = ROWS * DM;                 // 4M elements

    // 0) one-time weight transpose+convert, activation convert
    transpose_w<<<dim3(16, 16), dim3(32, 8), 0, stream>>>(Wq, Wtq);
    transpose_w<<<dim3(16, 16), dim3(32, 8), 0, stream>>>(Wk, Wtk);
    transpose_w<<<dim3(16, 16), dim3(32, 8), 0, stream>>>(Wv, Wtv);
    transpose_w<<<dim3(16, 16), dim3(32, 8), 0, stream>>>(Wo, Wto);
    f32_to_bf16<<<nAct / 4 / 256, blk256, 0, stream>>>(query, Qin, nAct);
    f32_to_bf16<<<nAct / 4 / 256, blk256, 0, stream>>>(key,   Kin, nAct);
    f32_to_bf16<<<nAct / 4 / 256, blk256, 0, stream>>>(value, Vin, nAct);

    // 1) projections -> bf16 head-major Q/K, transposed V
    gemm512_wmma<<<ROWS / 16, blk256, 0, stream>>>(Qin, Wtq, bq, nullptr, Qb, 0, ROWS);
    gemm512_wmma<<<ROWS / 16, blk256, 0, stream>>>(Kin, Wtk, bk, nullptr, Kb, 0, ROWS);
    gemm512_wmma<<<ROWS / 16, blk256, 0, stream>>>(Vin, Wtv, bv, nullptr, Vt, 1, ROWS);

    // 2) sparse sampling scores + top-40
    sample_scores<<<(BH * SEQ) / 256, blk256, 0, stream>>>(Qb, Kb, Msc);
    topk40<<<BH, blk256, 0, stream>>>(Msc, top);

    // 3) reduced attention via WMMA
    qk_scores_wmma<<<(BH * (SEQ / 16)) / 8, blk256, 0, stream>>>(Qb, Kb, top, scores);
    softmax_rows<<<(BH * RPAD * 32 + 255) / 256, blk256, 0, stream>>>(scores, attn);
    av_wmma<<<(BH * 12 * 32 + 255) / 256, blk256, 0, stream>>>(attn, Vt, outtop);

    // 4) context: mean(V) broadcast + scatter attended rows (bf16)
    vmean_kernel<<<(BH * DK * 32) / 256, blk256, 0, stream>>>(Vt, vmean);
    ctx_fill<<<(nAct / 4 + 255) / 256, blk256, 0, stream>>>(vmean, ctxb);
    ctx_scatter<<<(BH * NTOP * DK + 255) / 256, blk256, 0, stream>>>(top, outtop, ctxb);

    // 5) output projection -> d_out (fp32)
    gemm512_wmma<<<ROWS / 16, blk256, 0, stream>>>(ctxb, Wto, bo, out, nullptr, 0, ROWS);
}
